// CRF_39505109188855
// MI455X (gfx1250) — matrix operand order repaired and checked by reading, not compiled, not run on hardware
//
#include <hip/hip_runtime.h>
#include <hip/hip_bf16.h>
#include <stdint.h>

typedef __attribute__((ext_vector_type(16))) _Float16 v16h;
typedef __attribute__((ext_vector_type(8)))  float    v8f;

#define S_LEN 1024
#define NLAB  64
#define LTOT  66
#define BPG   16      // batch columns per workgroup (WMMA N)
#define LSTR  68      // LDS row stride for [b][j] tiles (16B-aligned, padded)

#define LOG2E 1.4426950408889634f
#define LN2   0.6931471805599453f

union Hv  { v16h v;    _Float16 e[16]; };
union Fv  { v8f  v;    float    f[8];  };
union Q16 { float4 q[4]; float  f[16]; };
union Q8  { float4 q[2]; float  f[8];  };

// async DMA: global -> LDS, 2x16B per thread (chunk and chunk+128B)
__device__ __forceinline__ void async_tile_b128x2(uint32_t lds_byte_addr,
                                                  const float* gaddr) {
  asm volatile("global_load_async_to_lds_b128 %0, %1, off\n\t"
               "global_load_async_to_lds_b128 %0, %1, off offset:128"
               :: "v"(lds_byte_addr), "v"(gaddr) : "memory");
}
__device__ __forceinline__ void wait_async_le2() {
  asm volatile("s_wait_asynccnt 0x2" ::: "memory");
}
__device__ __forceinline__ void wait_async_le0() {
  asm volatile("s_wait_asynccnt 0x0" ::: "memory");
}
__device__ __forceinline__ uint32_t lds_addr_of(const void* p) {
  return (uint32_t)(uintptr_t)p;   // flat LDS addr: low 32 bits = LDS offset
}

__global__ __launch_bounds__(128)
void crf_fwd_kernel(const float* __restrict__ logits,
                    const float* __restrict__ transition,
                    const int*   __restrict__ labels,
                    const int*   __restrict__ pmask,
                    float*       __restrict__ out)
{
  __shared__ __align__(16) float llds[3][BPG * LSTR];   // logit tile triple-buffer [b][j]
  __shared__ __align__(16) float Abuf[2][BPG * LSTR];   // alpha*log2e ping-pong [b][j]
  __shared__ __align__(16) float wmaxs[2][BPG * 4];     // per-wave column maxes [b][wave]
  __shared__ int   mlab[2][BPG];
  __shared__ int   mmsk[2][BPG];
  __shared__ float Tlds[LTOT * LTOT];

  const int tid  = threadIdx.x;
  const int wave = tid >> 5;
  const int lane = tid & 31;
  const int b    = lane & 15;   // column (N index)
  const int hlf  = lane >> 4;   // lane half
  const size_t bbase = (size_t)blockIdx.x * BPG;

  // ---- stage transition table into LDS ----
  for (int i = tid; i < LTOT * LTOT; i += 128) Tlds[i] = transition[i];

  // tile staging mapping: 16 rows x 64 floats, 128 threads x 2 float4
  const int rrow = tid >> 3;   // 0..15
  const int c0   = tid & 7;    // chunk 0..7 (covers c0 and c0+8)
  const float* lrow = logits + (bbase + rrow) * (size_t)S_LEN * NLAB;

  // per-slot loop-invariant addresses for this thread
  const uint32_t wsl[3] = { lds_addr_of(&llds[0][rrow * LSTR + c0 * 4]),
                            lds_addr_of(&llds[1][rrow * LSTR + c0 * 4]),
                            lds_addr_of(&llds[2][rrow * LSTR + c0 * 4]) };

  // stage tile t=0 synchronously
  {
    float4 t0a = *(const float4*)(lrow + c0 * 4);
    float4 t0b = *(const float4*)(lrow + c0 * 4 + 32);
    *(float4*)&llds[0][rrow * LSTR + c0 * 4]      = t0a;
    *(float4*)&llds[0][rrow * LSTR + c0 * 4 + 32] = t0b;
  }
  int labv = 0, mskv = 0;
  if (tid < 16)      { labv = labels[(bbase + tid) * S_LEN];        mlab[0][tid]      = labv; }
  else if (tid < 32) { mskv = pmask [(bbase + tid - 16) * S_LEN];   mmsk[0][tid - 16] = mskv; }

  // async-DMA tiles 1 and 2 straight into LDS
  async_tile_b128x2(wsl[1], lrow + 1 * NLAB + c0 * 4);
  async_tile_b128x2(wsl[2], lrow + 2 * NLAB + c0 * 4);

  // labels/mask prefetch for t=1
  if (tid < 16)      labv = labels[(bbase + tid) * S_LEN + 1];
  else if (tid < 32) mskv = pmask [(bbase + tid - 16) * S_LEN + 1];

  __syncthreads();

  // ---- A = exp(T) in WMMA A-layout registers (rows j = 16*wave + (lane&15)) ----
  Hv A0, A1;
  #pragma unroll
  for (int h = 0; h < 16; ++h) {
    int k = (h & 7) + 8 * hlf + 16 * (h >> 3);
    int j = 16 * wave + b;
    A0.e[h] = (_Float16)__builtin_amdgcn_exp2f(Tlds[j * LTOT + k]      * LOG2E);
    A1.e[h] = (_Float16)__builtin_amdgcn_exp2f(Tlds[j * LTOT + 32 + k] * LOG2E);
  }

  const int j0 = 16 * wave + 8 * hlf;   // 8 consecutive output rows per lane

  // ---- step 0: alpha1'[b][j] = log2e * (mask ? logit0[j] + T[j][start] : -100) ----
  {
    int mk = mmsk[0][b];
    Q8 LG, AN;
    LG.q[0] = *(const float4*)&llds[0][b * LSTR + j0];
    LG.q[1] = *(const float4*)&llds[0][b * LSTR + j0 + 4];
    float vmax = -3.0e38f;
    #pragma unroll
    for (int r = 0; r < 8; ++r) {
      int j = j0 + r;
      float upd = (LG.f[r] + Tlds[j * LTOT + 64]) * LOG2E;
      AN.f[r] = mk ? upd : (-100.0f * LOG2E);
      vmax = fmaxf(vmax, AN.f[r]);
    }
    *(float4*)&Abuf[1][b * LSTR + j0]     = AN.q[0];
    *(float4*)&Abuf[1][b * LSTR + j0 + 4] = AN.q[1];
    vmax = fmaxf(vmax, __shfl_xor(vmax, 16, 32));
    if (lane < 16) wmaxs[1][lane * 4 + wave] = vmax;
  }

  // gold-score streaming state (wave 0, lanes 0..15)
  float unary = 0.f, transacc = 0.f, prev_mkf = 0.f;
  int prev_lm = 64;
  if (wave == 0 && lane < 16) {
    int mk = mmsk[0][lane], lab = mlab[0][lane];
    float mkf = (float)mk;
    unary = mkf * llds[0][lane * LSTR + lab];
    int lmc = mk ? lab : 65;
    transacc = Tlds[lmc * LTOT + 64];   // s=0 term, weight 1 (from start)
    prev_lm = lmc; prev_mkf = mkf;
  }

  // commit lm for t=1, prefetch lm for t=2
  if (tid < 16)      mlab[1][tid]      = labv;
  else if (tid < 32) mmsk[1][tid - 16] = mskv;
  if (tid < 16)      labv = labels[(bbase + tid) * S_LEN + 2];
  else if (tid < 32) mskv = pmask [(bbase + tid - 16) * S_LEN + 2];

  wait_async_le2();    // tile 1 resident (tile 2 may still be in flight)
  __syncthreads();

  // One scan step; all buffer indices are compile-time literals.
  #define STEP(T_, CUR_, NXT_, CUR3_, WSLOT_, MORE_, TT_)                        \
  {                                                                              \
    /* commit staged labels/mask (T_+1), prefetch TT_ */                         \
    if (tid < 16)      mlab[NXT_][tid]      = labv;                              \
    else if (tid < 32) mmsk[NXT_][tid - 16] = mskv;                              \
    if (tid < 16)      labv = labels[(bbase + tid) * S_LEN + (TT_)];             \
    else if (tid < 32) mskv = pmask [(bbase + tid - 16) * S_LEN + (TT_)];        \
    if (MORE_)                                                                   \
      async_tile_b128x2(wsl[WSLOT_], lrow + (size_t)((T_) + 2) * NLAB + c0 * 4); \
    float4 wq = *(const float4*)&wmaxs[CUR_][b * 4];                             \
    float m2 = fmaxf(fmaxf(wq.x, wq.y), fmaxf(wq.z, wq.w));                      \
    const float* arow = &Abuf[CUR_][b * LSTR];                                   \
    Q16 QA, QB;                                                                  \
    _Pragma("unroll")                                                            \
    for (int c = 0; c < 4; ++c) {                                                \
      QA.q[c] = *(const float4*)(arow + 8 * hlf + 4 * (c & 1) + 16 * (c >> 1));  \
      QB.q[c] = *(const float4*)(arow + 32 + 8 * hlf + 4 * (c & 1) + 16 * (c >> 1)); \
    }                                                                            \
    Hv B0, B1;                                                                   \
    _Pragma("unroll")                                                            \
    for (int h = 0; h < 16; ++h) {                                               \
      B0.e[h] = (_Float16)__builtin_amdgcn_exp2f(QA.f[h] - m2);                  \
      B1.e[h] = (_Float16)__builtin_amdgcn_exp2f(QB.f[h] - m2);                  \
    }                                                                            \
    Fv C;                                                                        \
    _Pragma("unroll")                                                            \
    for (int i = 0; i < 8; ++i) C.f[i] = 0.f;                                    \
    C.v = __builtin_amdgcn_wmma_f32_16x16x32_f16(false, A0.v, false, B0.v,       \
                                                 (short)0, C.v, false, false);   \
    C.v = __builtin_amdgcn_wmma_f32_16x16x32_f16(false, A1.v, false, B1.v,       \
                                                 (short)0, C.v, false, false);   \
    int mk = mmsk[CUR_][b];                                                      \
    Q8 LG, AO, AN;                                                               \
    LG.q[0] = *(const float4*)&llds[CUR3_][b * LSTR + j0];                       \
    LG.q[1] = *(const float4*)&llds[CUR3_][b * LSTR + j0 + 4];                   \
    AO.q[0] = *(const float4*)(arow + j0);                                       \
    AO.q[1] = *(const float4*)(arow + j0 + 4);                                   \
    float vmax = -3.0e38f;                                                       \
    _Pragma("unroll")                                                            \
    for (int r = 0; r < 8; ++r) {                                                \
      float tr  = __builtin_amdgcn_logf(C.f[r]) + m2;                            \
      float upd = fmaf(LG.f[r], LOG2E, tr);                                      \
      AN.f[r] = mk ? upd : AO.f[r];                                              \
      vmax = fmaxf(vmax, AN.f[r]);                                               \
    }                                                                            \
    *(float4*)&Abuf[NXT_][b * LSTR + j0]     = AN.q[0];                          \
    *(float4*)&Abuf[NXT_][b * LSTR + j0 + 4] = AN.q[1];                          \
    vmax = fmaxf(vmax, __shfl_xor(vmax, 16, 32));                                \
    if (lane < 16) wmaxs[NXT_][lane * 4 + wave] = vmax;                          \
    if (wave == 0 && lane < 16) {                                                \
      int mk2 = mmsk[CUR_][lane], lab = mlab[CUR_][lane];                        \
      float mkf = (float)mk2;                                                    \
      unary += mkf * llds[CUR3_][lane * LSTR + lab];                             \
      int lmc = mk2 ? lab : 65;                                                  \
      transacc += Tlds[lmc * LTOT + prev_lm] * prev_mkf;                         \
      prev_lm = lmc; prev_mkf = mkf;                                             \
    }                                                                            \
    if (MORE_) wait_async_le2(); else wait_async_le0();                          \
    __syncthreads();                                                             \
  }

  // ============ main scan, unrolled x6 (period lcm(2,3)) ============
  // t = tb+u, tb = 1+6k:  cur=(t&1), cur3=(t%3), write slot=((t+2)%3)
  #pragma clang loop unroll(disable)
  for (int tb = 1; tb <= S_LEN - 9; tb += 6) {
    STEP(tb + 0, 1, 0, 1, 0, 1, tb + 2)
    STEP(tb + 1, 0, 1, 2, 1, 1, tb + 3)
    STEP(tb + 2, 1, 0, 0, 2, 1, tb + 4)
    STEP(tb + 3, 0, 1, 1, 0, 1, tb + 5)
    STEP(tb + 4, 1, 0, 2, 1, 1, tb + 6)
    STEP(tb + 5, 0, 1, 0, 2, 1, tb + 7)
  }
  // tail: t = 1021, 1022, 1023 (prefetch shuts off, tt clamps)
  STEP(1021, 1, 0, 1, 0, 1, 1023)
  STEP(1022, 0, 1, 2, 1, 0, 1023)
  STEP(1023, 1, 0, 0, 2, 0, 1023)
  #undef STEP

  // ================= epilogue: norm + output =================
  if (wave == 0 && lane < 16) {
    transacc += Tlds[65 * LTOT + prev_lm] * prev_mkf;    // s=S term (to end)
    const int fin = 0;                                   // final Abuf written by t=1023
    float m2 = -3.0e38f;
    for (int j = 0; j < NLAB; ++j)
      m2 = fmaxf(m2, Abuf[fin][lane * LSTR + j] * LN2 + Tlds[65 * LTOT + j]);
    float ssum = 0.f;
    for (int j = 0; j < NLAB; ++j)
      ssum += __builtin_amdgcn_exp2f(
          (Abuf[fin][lane * LSTR + j] * LN2 + Tlds[65 * LTOT + j] - m2) * LOG2E);
    float norm = m2 + __builtin_amdgcn_logf(ssum) * LN2;
    out[bbase + lane] = norm - (unary + transacc);
  }
}

extern "C" void kernel_launch(void* const* d_in, const int* in_sizes, int n_in,
                              void* d_out, int out_size, void* d_ws, size_t ws_size,
                              hipStream_t stream) {
  const float* logits     = (const float*)d_in[0];
  const float* transition = (const float*)d_in[1];
  const int*   labels     = (const int*)d_in[2];
  const int*   pmask      = (const int*)d_in[3];
  float* out = (float*)d_out;

  const int B = in_sizes[2] / S_LEN;   // labels is [B,S]
  dim3 grid(B / BPG), block(128);
  hipLaunchKernelGGL(crf_fwd_kernel, grid, block, 0, stream,
                     logits, transition, labels, pmask, out);
}
